// CrossAttentionTest_11793980195384
// MI455X (gfx1250) — matrix operand order repaired and checked by reading, not compiled, not run on hardware
//
#include <hip/hip_runtime.h>
#include <hip/hip_bf16.h>

typedef __attribute__((ext_vector_type(16))) _Float16 v16h;
typedef __attribute__((ext_vector_type(8)))  float    v8f;
typedef __attribute__((ext_vector_type(4)))  int      v4i;

#define NB   256   // batches
#define NS   256   // seq
#define ND   256   // model dim
#define NH   4     // heads
#define NDH  64    // head dim

typedef __attribute__((address_space(1))) v4i* g128_t;   // global b128
typedef __attribute__((address_space(3))) v4i* l128_t;   // LDS b128

// ---------------------------------------------------------------------------
// Async global->LDS 16-byte copy (GLOBAL_LOAD_ASYNC_TO_LDS_B128, ASYNCcnt).
__device__ __forceinline__ void async_cp16(const void* g, void* l) {
#if __has_builtin(__builtin_amdgcn_global_load_async_to_lds_b128)
    __builtin_amdgcn_global_load_async_to_lds_b128((g128_t)g, (l128_t)l, 0, 0);
#else
    unsigned loff = (unsigned)(unsigned long long)(l128_t)l;
    asm volatile("global_load_async_to_lds_b128 %0, %1, off"
                 :: "v"(loff), "v"((unsigned long long)g) : "memory");
#endif
}

template <int N>
__device__ __forceinline__ void wait_async() {
#if __has_builtin(__builtin_amdgcn_s_wait_asynccnt)
    __builtin_amdgcn_s_wait_asynccnt(N);
#else
    asm volatile("s_wait_asynccnt %0" :: "n"(N) : "memory");
#endif
}

// ---------------------------------------------------------------------------
// WMMA f16 operand gathers from LDS tiles (layouts per CDNA5 ISA 7.12.2).
// A operand: lane l holds row M=l&15; element e -> K=(e<8?e:e+8)+8*(l>>4)
__device__ __forceinline__ v16h ld_opA(const _Float16* t, int ld, int lane) {
    const _Float16* p = t + (lane & 15) * ld + ((lane >> 4) << 3);
    union { v16h h; unsigned u[8]; } r;
    r.u[0] = *(const unsigned*)(p + 0);
    r.u[1] = *(const unsigned*)(p + 2);
    r.u[2] = *(const unsigned*)(p + 4);
    r.u[3] = *(const unsigned*)(p + 6);
    r.u[4] = *(const unsigned*)(p + 16);
    r.u[5] = *(const unsigned*)(p + 18);
    r.u[6] = *(const unsigned*)(p + 20);
    r.u[7] = *(const unsigned*)(p + 22);
    return r.h;
}
// B operand (N-major rows): lane l holds col N=l&15; element e -> K=(e<8?e:e+8)+8*(l>>4)
__device__ __forceinline__ v16h ld_opB(const _Float16* t, int ld, int lane) {
    const _Float16* p = t + (lane & 15) * ld + ((lane >> 4) << 3);
    union { v16h h; unsigned u[8]; } r;
#pragma unroll
    for (int i = 0; i < 8; ++i) r.u[i] = *(const unsigned*)(p + 2 * i + (i >= 4 ? 8 : 0));
    return r.h;
}

// ---------------------------------------------------------------------------
// Kernel 1: per-batch group rank & size. One block of 256 threads.
__global__ void group_kernel(const int* __restrict__ ids,
                             int* __restrict__ rank, int* __restrict__ gsz) {
    int b = threadIdx.x;
    int id = ids[b], r = 0, g = 0;
    for (int j = 0; j < NB; ++j) {
        bool s = (ids[j] == id);
        g += s;
        r += (s && (j < b));
    }
    rank[b] = r;
    gsz[b]  = g;
}

// Kernel 2: weights -> f16, transposed to N-major. 4*256*256 elems.
__global__ void prep_weights_kernel(const float* __restrict__ Wq,
                                    const float* __restrict__ Wk,
                                    const float* __restrict__ Wv,
                                    const float* __restrict__ Wo,
                                    _Float16* __restrict__ Wt) {
    int i = blockIdx.x * 256 + threadIdx.x;      // 0 .. 4*65536-1
    int mat = i >> 16, rest = i & 65535;
    int n = rest >> 8, k = rest & 255;
    const float* W = (mat == 0) ? Wq : (mat == 1) ? Wk : (mat == 2) ? Wv : Wo;
    Wt[(size_t)mat * 65536 + n * ND + k] = (_Float16)W[k * ND + n];
}

// Kernel 3: enh = x + scale*obj_emb[rank[b]], cast to f16.
__global__ void enh_kernel(const float* __restrict__ x,
                           const float* __restrict__ obj,
                           const float* __restrict__ scale,
                           const int* __restrict__ rank,
                           _Float16* __restrict__ out) {
    size_t i = (size_t)blockIdx.x * 256 + threadIdx.x;   // B*S*D
    int d = (int)(i & 255);
    int b = (int)(i >> 16);
    out[i] = (_Float16)(x[i] + scale[0] * obj[rank[b] * ND + d]);
}

// ---------------------------------------------------------------------------
// Kernel 4: C[M=B*S][N=256] = A[f16] @ W + bias, WMMA f16->f32.
// 256 thr (8 waves); block tile 128 rows x 256 cols; async double-buffered
// LDS staging (6 GLOBAL_LOAD_ASYNC_TO_LDS_B128 per thread per 32-K slice).
// MODE 0: f16 row-major (Q,K). MODE 1: f16 Vt[b][n][s]. MODE 2: f32
// out = gsize>1 ? resid + C : resid.
template <int MODE>
__global__ __launch_bounds__(256) void gemm_kernel(
    const _Float16* __restrict__ A,    // [B*S][256]
    const _Float16* __restrict__ Wt,   // [256 N][256 K] N-major f16
    const float*    __restrict__ bias, // [256]
    _Float16*       __restrict__ outH,
    float*          __restrict__ outF,
    const float*    __restrict__ resid,
    const int*      __restrict__ gsz) {
    __shared__ _Float16 sA[2][128 * 32];   // 2 x 8 KB
    __shared__ _Float16 sB[2][256 * 32];   // 2 x 16 KB
    const int tid = threadIdx.x, lane = tid & 31, wv = tid >> 5;
    const int row0 = blockIdx.x * 128;

    v8f acc[16];
#pragma unroll
    for (int i = 0; i < 16; ++i) acc[i] = (v8f){};

    const int rA = tid >> 1, phA = (tid & 1) * 16;     // A: 32-halves half-row
    // stage slice 'sl' (k0 = 32*sl) into buffer 'buf': 6 async b128 per thread
#define STAGE_SLICE(sl, buf)                                                   \
    do {                                                                       \
        int k0_ = (sl) * 32;                                                   \
        const _Float16* sa = A + (size_t)(row0 + rA) * ND + k0_ + phA;         \
        _Float16* da = sA[buf] + rA * 32 + phA;                                \
        async_cp16(sa, da);                                                    \
        async_cp16(sa + 8, da + 8);                                            \
        const _Float16* sb = Wt + (size_t)tid * ND + k0_;                      \
        _Float16* db = sB[buf] + tid * 32;                                     \
        async_cp16(sb + 0,  db + 0);                                           \
        async_cp16(sb + 8,  db + 8);                                           \
        async_cp16(sb + 16, db + 16);                                          \
        async_cp16(sb + 24, db + 24);                                          \
    } while (0)

    STAGE_SLICE(0, 0);
    for (int s = 0; s < 8; ++s) {
        const int buf = s & 1;
        if (s < 7) {
            STAGE_SLICE(s + 1, buf ^ 1);
            wait_async<6>();            // slice s landed; s+1 stays in flight
        } else {
            wait_async<0>();
        }
        __syncthreads();

        v16h a = ld_opA(sA[buf] + wv * 16 * 32, 32, lane);
        v16h bcur = ld_opB(sB[buf], 32, lane);
#pragma unroll
        for (int nt = 0; nt < 16; ++nt) {
            v16h bnext;
            if (nt < 15) bnext = ld_opB(sB[buf] + (nt + 1) * 16 * 32, 32, lane);
            acc[nt] = __builtin_amdgcn_wmma_f32_16x16x32_f16(
                false, a, false, bcur, (short)0, acc[nt], false, false);
            if (nt < 15) bcur = bnext;
        }
        __syncthreads();                // buf may be overwritten next iter
    }
#undef STAGE_SLICE

    const int mloc = (lane >> 4) * 8, n0 = lane & 15;
#pragma unroll
    for (int nt = 0; nt < 16; ++nt) {
        int n = nt * 16 + n0;
        float bn = bias[n];
#pragma unroll
        for (int r = 0; r < 8; ++r) {
            int m = row0 + wv * 16 + mloc + r;
            float v = acc[nt][r] + bn;
            if (MODE == 0) {
                outH[(size_t)m * ND + n] = (_Float16)v;
            } else if (MODE == 1) {
                int b = m >> 8, sq = m & 255;         // Vt[b][n][s]
                outH[(size_t)b * 65536 + (size_t)n * NS + sq] = (_Float16)v;
            } else {
                size_t idx = (size_t)m * ND + n;
                float base = resid[idx];
                outF[idx] = (gsz[m >> 8] > 1) ? base + v : base;
            }
        }
    }
}

// ---------------------------------------------------------------------------
// Kernel 5: attention for one (batch, head, 128-q-row block).
// grid.x = B*H*2. 8 waves; wave owns 16 q rows (full 256-key score strip).
__global__ __launch_bounds__(256) void attn_kernel(
    const _Float16* __restrict__ Q,    // [B*S][256]
    const _Float16* __restrict__ Km,   // [B*S][256]
    const _Float16* __restrict__ Vt,   // [B][256 d][256 s]
    _Float16*       __restrict__ Ctx)  // [B*S][256]
{
    __shared__ _Float16 sQ[128 * 64];       // 16 KB
    __shared__ _Float16 sK[256 * 64];       // 32 KB
    __shared__ _Float16 sV[64 * 256];       // 32 KB
    __shared__ _Float16 sP[8 * 16 * 128];   // 32 KB (wave-private prob tiles)

    const int bx = blockIdx.x;
    const int b = bx >> 3, h = (bx >> 1) & 3, qb = bx & 1;
    const int q0 = qb * 128;
    const int tid = threadIdx.x, lane = tid & 31, wv = tid >> 5;

    {   // stage Q strip: 128 rows x 64 halves (4 async b128 / thread)
        int r = tid >> 1, ph = (tid & 1) * 32;
        const _Float16* src = Q + (size_t)(b * NS + q0 + r) * ND + h * NDH + ph;
        _Float16* dst = sQ + r * 64 + ph;
#pragma unroll
        for (int i = 0; i < 4; ++i) async_cp16(src + 8 * i, dst + 8 * i);
    }
    {   // stage all keys: 256 rows x 64 halves (8 async b128 / thread)
        const _Float16* src = Km + (size_t)(b * NS + tid) * ND + h * NDH;
        _Float16* dst = sK + tid * 64;
#pragma unroll
        for (int i = 0; i < 8; ++i) async_cp16(src + 8 * i, dst + 8 * i);
    }
    {   // stage Vt head slab: 64 d-rows x 256 halves (8 async b128 / thread)
        int r = tid >> 2, ph = (tid & 3) * 64;
        const _Float16* src = Vt + (size_t)(b * ND + h * NDH + r) * NS + ph;
        _Float16* dst = sV + r * 256 + ph;
#pragma unroll
        for (int i = 0; i < 8; ++i) async_cp16(src + 8 * i, dst + 8 * i);
    }
    wait_async<0>();
    __syncthreads();

    // ---- scores: 16 q rows x 256 keys per wave, f32 accum
    v8f sc[16];
#pragma unroll
    for (int i = 0; i < 16; ++i) sc[i] = (v8f){};
#pragma unroll
    for (int kk = 0; kk < 2; ++kk) {
        v16h a = ld_opA(sQ + wv * 16 * 64 + kk * 32, 64, lane);
        v16h bcur = ld_opB(sK + kk * 32, 64, lane);
#pragma unroll
        for (int nt = 0; nt < 16; ++nt) {
            v16h bnext;
            if (nt < 15) bnext = ld_opB(sK + (nt + 1) * 16 * 64 + kk * 32, 64, lane);
            sc[nt] = __builtin_amdgcn_wmma_f32_16x16x32_f16(
                false, a, false, bcur, (short)0, sc[nt], false, false);
            if (nt < 15) bcur = bnext;
        }
    }

    // ---- softmax over keys (rows live in element r; lanes of a 16-lane
    // half hold distinct keys -> shfl_xor 1,2,4,8 reduces within the half)
    const float scl = 0.125f;   // 1/sqrt(64)
#pragma unroll
    for (int r = 0; r < 8; ++r) {
        float mx = -3.0e38f;
#pragma unroll
        for (int t = 0; t < 16; ++t) mx = fmaxf(mx, sc[t][r]);
#pragma unroll
        for (int off = 1; off < 16; off <<= 1) mx = fmaxf(mx, __shfl_xor(mx, off, 32));
        float sum = 0.f;
#pragma unroll
        for (int t = 0; t < 16; ++t) {
            float e = __expf((sc[t][r] - mx) * scl);
            sc[t][r] = e;
            sum += e;
        }
#pragma unroll
        for (int off = 1; off < 16; off <<= 1) sum += __shfl_xor(sum, off, 32);
        float inv = 1.0f / sum;
#pragma unroll
        for (int t = 0; t < 16; ++t) sc[t][r] *= inv;
    }

    // ---- ctx = P @ V (keys in two 128-wide passes through wave-private LDS)
    v8f cacc[4];
#pragma unroll
    for (int i = 0; i < 4; ++i) cacc[i] = (v8f){};
    _Float16* myP = sP + wv * 16 * 128;
#pragma unroll
    for (int half = 0; half < 2; ++half) {
#pragma unroll
        for (int t = 0; t < 8; ++t) {
            int tt = half * 8 + t;
#pragma unroll
            for (int r = 0; r < 8; ++r) {
                int M = (lane >> 4) * 8 + r;
                myP[M * 128 + t * 16 + (lane & 15)] = (_Float16)sc[tt][r];
            }
        }
        // same-wave LDS ops are in-order: stores precede the gathers below
#pragma unroll
        for (int kb = 0; kb < 4; ++kb) {
            v16h a = ld_opA(myP + kb * 32, 128, lane);
#pragma unroll
            for (int nt = 0; nt < 4; ++nt) {
                v16h bb = ld_opB(sV + nt * 16 * 256 + half * 128 + kb * 32, 256, lane);
                cacc[nt] = __builtin_amdgcn_wmma_f32_16x16x32_f16(
                    false, a, false, bb, (short)0, cacc[nt], false, false);
            }
        }
    }

    // ---- store ctx f16: col = h*64 + 16*nt + (lane&15)
#pragma unroll
    for (int nt = 0; nt < 4; ++nt) {
        int d = h * NDH + nt * 16 + (lane & 15);
#pragma unroll
        for (int r = 0; r < 8; ++r) {
            int s = q0 + wv * 16 + (lane >> 4) * 8 + r;
            Ctx[(size_t)(b * NS + s) * ND + d] = (_Float16)cacc[nt][r];
        }
    }
}

// ---------------------------------------------------------------------------
extern "C" void kernel_launch(void* const* d_in, const int* in_sizes, int n_in,
                              void* d_out, int out_size, void* d_ws, size_t ws_size,
                              hipStream_t stream) {
    const float* batch_seq = (const float*)d_in[0];
    const int*   img_ids   = (const int*)  d_in[1];
    const float* Wq = (const float*)d_in[2];
    const float* Wk = (const float*)d_in[3];
    const float* Wv = (const float*)d_in[4];
    const float* Wo = (const float*)d_in[5];
    const float* bq = (const float*)d_in[6];
    const float* bk = (const float*)d_in[7];
    const float* bv = (const float*)d_in[8];
    const float* bo = (const float*)d_in[9];
    const float* obj_emb = (const float*)d_in[10];
    const float* scale   = (const float*)d_in[11];
    float* out = (float*)d_out;

    char* ws = (char*)d_ws;
    int*      rank  = (int*)ws;                         // 256
    int*      gsz   = rank + 256;                       // 256
    _Float16* WtAll = (_Float16*)(ws + 4096);           // 4*65536 f16
    const size_t MB32 = (size_t)1 << 24;                // elems per f16 buffer
    _Float16* enh = (_Float16*)(ws + (1 << 20));        // also reused as ctx
    _Float16* Qb  = enh + MB32;
    _Float16* Kb  = Qb  + MB32;
    _Float16* Vtb = Kb  + MB32;

    group_kernel<<<1, 256, 0, stream>>>(img_ids, rank, gsz);
    prep_weights_kernel<<<1024, 256, 0, stream>>>(Wq, Wk, Wv, Wo, WtAll);
    enh_kernel<<<65536, 256, 0, stream>>>(batch_seq, obj_emb, scale, rank, enh);

    gemm_kernel<0><<<512, 256, 0, stream>>>(enh, WtAll,             bq, Qb,  nullptr, nullptr, nullptr);
    gemm_kernel<0><<<512, 256, 0, stream>>>(enh, WtAll + 1 * 65536, bk, Kb,  nullptr, nullptr, nullptr);
    gemm_kernel<1><<<512, 256, 0, stream>>>(enh, WtAll + 2 * 65536, bv, Vtb, nullptr, nullptr, nullptr);

    attn_kernel<<<NB * NH * 2, 256, 0, stream>>>(Qb, Kb, Vtb, enh /* ctx reuse */);

    gemm_kernel<2><<<512, 256, 0, stream>>>(enh, WtAll + 3 * 65536, bo, nullptr,
                                            out, batch_seq, gsz);
}
